// GATNNA_88510686036345
// MI455X (gfx1250) — compile-verified, matrix operand order checked by reading, not run on hardware
//
#include <hip/hip_runtime.h>
#include <hip/hip_bf16.h>
#include <math.h>
#include <stdint.h>

// ---- problem sizes (fixed by the reference) ----
#define BB    32
#define NNODE 512
#define WDIM  128
#define EDIM  256
#define HEADS 8
#define DK    64
#define HD    512              // HEADS*DK
#define MROWS 16384            // BB*NNODE
#define KDOWN 131072           // NNODE*EDIM
#define ODIM  256
#define AWORDS (NNODE / 32)    // 16 mask words per row

// CDNA5 async global->LDS path (global_load_async_to_lds_b128 + s_wait_asynccnt).
#define USE_ASYNC_LDS 1

typedef __bf16 bf16;
typedef __attribute__((ext_vector_type(16))) __bf16 bf16x16;
typedef __attribute__((ext_vector_type(8)))  float  f32x8;

// ---------------- small device helpers ----------------
__device__ __forceinline__ bf16 f2bf(float f) {
  union { float f; unsigned u; } cv; cv.f = f;
  unsigned r = cv.u + 0x7FFFu + ((cv.u >> 16) & 1u);   // RNE
  union { unsigned short u; bf16 b; } o; o.u = (unsigned short)(r >> 16);
  return o.b;
}

__device__ __forceinline__ f32x8 wmma_bf16(bf16x16 a, bf16x16 b, f32x8 c) {
  return __builtin_amdgcn_wmma_f32_16x16x32_bf16(false, a, false, b, (short)0, c, false, false);
}

// 16x32 fragment from a row-major source (used for A, K^T, P, and all
// transposed-weight B operands): lane&15 -> row, (lane>>4)*8 -> K half-group,
// two contiguous 8-element (16B) chunks -> ds_load_b128 / global_load_b128.
__device__ __forceinline__ bf16x16 fragA(const bf16* p, int ld, int lane) {
  const bf16* q = p + (long)(lane & 15) * ld + ((lane >> 4) << 3);
  bf16x16 r;
  *(uint4*)&r         = *(const uint4*)q;         // elems 0..7
  *(((uint4*)&r) + 1) = *(const uint4*)(q + 16);  // elems 8..15
  return r;
}

__device__ __forceinline__ float hmax16(float v) {
#pragma unroll
  for (int m = 1; m < 16; m <<= 1) v = fmaxf(v, __shfl_xor(v, m, 32));
  return v;
}
__device__ __forceinline__ float hsum16(float v) {
#pragma unroll
  for (int m = 1; m < 16; m <<= 1) v += __shfl_xor(v, m, 32);
  return v;
}

// async (or fallback) 16-byte global -> LDS copy
__device__ __forceinline__ void g2lds16(bf16* lds, const bf16* g) {
#if USE_ASYNC_LDS
  asm volatile("global_load_async_to_lds_b128 %0, %1, off"
               :: "v"((unsigned)(uintptr_t)lds), "v"(g) : "memory");
#else
  *(uint4*)lds = *(const uint4*)g;
#endif
}
__device__ __forceinline__ void wait_async() {
#if USE_ASYNC_LDS
  asm volatile("s_wait_asynccnt 0x0" ::: "memory");
#endif
}

// ---------------- kernels ----------------
__global__ __launch_bounds__(256) void k_cvt(const float* __restrict__ s,
                                             bf16* __restrict__ d, long n) {
  long i = (long)blockIdx.x * blockDim.x + threadIdx.x;
  long stride = (long)gridDim.x * blockDim.x;
  for (; i < n; i += stride) d[i] = f2bf(s[i]);
}

// pack 0/1 adjacency into bit-words: m[row*AWORDS + w] bit j = (adj[row*N + w*32+j] != 0)
__global__ __launch_bounds__(256) void k_adjpack(const int* __restrict__ adj,
                                                 unsigned* __restrict__ m, long words) {
  long i = (long)blockIdx.x * 256 + threadIdx.x;
  if (i >= words) return;
  const int* p = adj + i * 32;
  unsigned w = 0;
#pragma unroll
  for (int j = 0; j < 32; j++) w |= (p[j] != 0 ? 1u : 0u) << j;
  m[i] = w;
}

// transpose + convert: d[n*K + k] = s[k*N + n].  grid (K/32, N/32), block (32,8)
__global__ void k_cvt_t(const float* __restrict__ s, bf16* __restrict__ d,
                        int K, int N) {
  __shared__ bf16 t[32][33];
  const int k0 = blockIdx.x * 32, n0 = blockIdx.y * 32;
  const int tx = threadIdx.x, ty = threadIdx.y;
#pragma unroll
  for (int j = 0; j < 4; j++)
    t[ty + j * 8][tx] = f2bf(s[(long)(k0 + ty + j * 8) * N + n0 + tx]);
  __syncthreads();
#pragma unroll
  for (int j = 0; j < 4; j++)
    d[(long)(n0 + ty + j * 8) * K + k0 + tx] = t[tx][ty + j * 8];
}

// per-batch transpose of V: vt[(b*HD + d)*NNODE + n] = vb[(b*NNODE + n)*HD + d]
// grid (NNODE/32, HD/32, BB), block (32,8)
__global__ void k_vtrans(const bf16* __restrict__ s, bf16* __restrict__ d) {
  __shared__ bf16 t[32][33];
  const int b = blockIdx.z;
  const int n0 = blockIdx.x * 32, d0 = blockIdx.y * 32;
  const int tx = threadIdx.x, ty = threadIdx.y;
#pragma unroll
  for (int j = 0; j < 4; j++)
    t[ty + j * 8][tx] = s[((long)b * NNODE + n0 + ty + j * 8) * HD + d0 + tx];
  __syncthreads();
#pragma unroll
  for (int j = 0; j < 4; j++)
    d[((long)b * HD + d0 + ty + j * 8) * NNODE + n0 + tx] = t[tx][ty + j * 8];
}

// C[M,N] = A[M,K] * Bt[N,K]^T + bias.  Block tile 128x128, K-step 32,
// 8 waves each computing 32x64.  A staged in LDS (async, double-buffered);
// B fragments loaded straight from the transposed weight in global/L2.
template <bool RELU, bool OUTF, bool OUTB>
__global__ __launch_bounds__(256) void k_gemm(const bf16* __restrict__ A,
                                              const bf16* __restrict__ Bt,
                                              const float* __restrict__ bias,
                                              float* __restrict__ outF,
                                              bf16* __restrict__ outB,
                                              int M, int Ncol, int K) {
  __shared__ bf16 As[2][128][40];
  const int tid = threadIdx.x, lane = tid & 31, wid = tid >> 5;
  const int half = (lane >> 4) << 3, nn = lane & 15;
  const int bm = blockIdx.x * 128, bn = blockIdx.y * 128;
  const int wm = (wid & 3) * 32, wn = (wid >> 2) * 64;
  f32x8 acc[2][4] = {};

  // prologue: stage first A tile
#pragma unroll
  for (int it = 0; it < 2; it++) {
    int idx = tid * 8 + it * 2048;
    int r = idx >> 5, c = idx & 31;
    g2lds16(&As[0][r][c], &A[(long)(bm + r) * K + c]);
  }

  for (int k0 = 0, pb = 0; k0 < K; k0 += 32, pb ^= 1) {
    wait_async();
    __syncthreads();
    if (k0 + 32 < K) {
#pragma unroll
      for (int it = 0; it < 2; it++) {
        int idx = tid * 8 + it * 2048;
        int r = idx >> 5, c = idx & 31;
        g2lds16(&As[pb ^ 1][r][c], &A[(long)(bm + r) * K + k0 + 32 + c]);
      }
    }
    bf16x16 bfr[4];
#pragma unroll
    for (int ni = 0; ni < 4; ni++)
      bfr[ni] = fragA(Bt + (long)(bn + wn + ni * 16) * K + k0, K, lane);
#pragma unroll
    for (int mi = 0; mi < 2; mi++) {
      bf16x16 afr = fragA(&As[pb][wm + mi * 16][0], 40, lane);
#pragma unroll
      for (int ni = 0; ni < 4; ni++)
        acc[mi][ni] = wmma_bf16(afr, bfr[ni], acc[mi][ni]);
    }
  }

#pragma unroll
  for (int mi = 0; mi < 2; mi++) {
#pragma unroll
    for (int ni = 0; ni < 4; ni++) {
      int col = bn + wn + ni * 16 + nn;
      float bia = bias[col];
#pragma unroll
      for (int r = 0; r < 8; r++) {
        float vv = acc[mi][ni][r] + bia;
        if (RELU) vv = fmaxf(vv, 0.f);
        long row = (long)(bm + wm + mi * 16 + half + r);
        if (OUTF) outF[row * Ncol + col] = vv;
        if (OUTB) outB[row * Ncol + col] = f2bf(vv);
      }
    }
  }
}

// Flash-style masked attention. grid = (NNODE/128, BB*HEADS). Each wave: 16 q rows.
// q,k: [B*N, HD] row-major ; vt: [B][HD][N] (pre-transposed) ;
// adjm: packed bit-mask [B*N][AWORDS] ; ctx: [B*N, HD].
__global__ __launch_bounds__(256) void k_attn(const bf16* __restrict__ q,
                                              const bf16* __restrict__ k,
                                              const bf16* __restrict__ vt,
                                              const unsigned* __restrict__ adjm,
                                              bf16* __restrict__ ctx) {
  __shared__ bf16 Pbuf[8][16][40];
  const int tid = threadIdx.x, lane = tid & 31, wid = tid >> 5;
  const int half = (lane >> 4) << 3, nn = lane & 15;
  const int bh = blockIdx.y, bb = bh >> 3, h = bh & 7;
  const int q0 = blockIdx.x * 128 + wid * 16;
  const long rowQ = (long)bb * NNODE + q0;
  bf16x16 qf0 = fragA(q + rowQ * HD + h * DK,      HD, lane);
  bf16x16 qf1 = fragA(q + rowQ * HD + h * DK + 32, HD, lane);
  f32x8 o[4] = {};
  float mrow[8], lrow[8];
#pragma unroll
  for (int r = 0; r < 8; r++) { mrow[r] = -INFINITY; lrow[r] = 0.f; }

  for (int kc = 0; kc < NNODE; kc += 32) {
    f32x8 s[2];
#pragma unroll
    for (int t = 0; t < 2; t++) {
      long rowK = (long)bb * NNODE + kc + t * 16;
      bf16x16 kf0 = fragA(k + rowK * HD + h * DK,      HD, lane);
      bf16x16 kf1 = fragA(k + rowK * HD + h * DK + 32, HD, lane);
      f32x8 z = {};
      s[t] = wmma_bf16(qf0, kf0, z);
      s[t] = wmma_bf16(qf1, kf1, s[t]);
    }
#pragma unroll
    for (int r = 0; r < 8; r++) {
      // one packed word covers this row's whole 32-key chunk
      unsigned w = adjm[((long)bb * NNODE + q0 + half + r) * AWORDS + (kc >> 5)];
      float s0 = ((w >> nn)        & 1u) ? s[0][r] * 0.125f : -1e9f;
      float s1 = ((w >> (16 + nn)) & 1u) ? s[1][r] * 0.125f : -1e9f;
      float mnew = fmaxf(mrow[r], hmax16(fmaxf(s0, s1)));
      float resc = __expf(mrow[r] - mnew);
      mrow[r] = mnew;
      float p0 = __expf(s0 - mnew), p1 = __expf(s1 - mnew);
      lrow[r] = lrow[r] * resc + hsum16(p0 + p1);
#pragma unroll
      for (int ni = 0; ni < 4; ni++) o[ni][r] *= resc;
      Pbuf[wid][half + r][nn]      = f2bf(p0);
      Pbuf[wid][half + r][16 + nn] = f2bf(p1);
    }
    bf16x16 pf = fragA(&Pbuf[wid][0][0], 40, lane);
#pragma unroll
    for (int ni = 0; ni < 4; ni++) {
      bf16x16 vf = fragA(vt + ((long)bb * HD + h * DK + ni * 16) * NNODE + kc,
                         NNODE, lane);
      o[ni] = wmma_bf16(pf, vf, o[ni]);
    }
  }
#pragma unroll
  for (int r = 0; r < 8; r++) {
    float inv = 1.f / lrow[r];
    long row = (long)bb * NNODE + q0 + half + r;
#pragma unroll
    for (int ni = 0; ni < 4; ni++)
      ctx[row * HD + h * DK + ni * 16 + nn] = f2bf(o[ni][r] * inv);
  }
}

// residual add + LayerNorm over E=256; one block per row.
__global__ __launch_bounds__(256) void k_addln(const float* __restrict__ x,
                                               const float* __restrict__ res,
                                               const float* __restrict__ g,
                                               const float* __restrict__ bta,
                                               float* __restrict__ outF,
                                               bf16* __restrict__ outB) {
  __shared__ float red[256];
  const int row = blockIdx.x, t = threadIdx.x;
  float v = x[(long)row * EDIM + t] + res[(long)row * EDIM + t];
  red[t] = v; __syncthreads();
  for (int s = 128; s > 0; s >>= 1) { if (t < s) red[t] += red[t + s]; __syncthreads(); }
  float mu = red[0] / EDIM; __syncthreads();
  float d = v - mu;
  red[t] = d * d; __syncthreads();
  for (int s = 128; s > 0; s >>= 1) { if (t < s) red[t] += red[t + s]; __syncthreads(); }
  float var = red[0] / EDIM;
  float y = d * rsqrtf(var + 1e-5f) * g[t] + bta[t];
  if (outF) outF[(long)row * EDIM + t] = y;
  if (outB) outB[(long)row * EDIM + t] = f2bf(y);
}

// down pass1: K-split WMMA partials; Wdt is [ODIM][KDOWN] (transposed).
// grid = (ODIM/16, 64).
__global__ __launch_bounds__(256) void k_down1(const bf16* __restrict__ A,
                                               const bf16* __restrict__ Wdt,
                                               float* __restrict__ part) {
  __shared__ float red[8][2][16][16];
  const int tid = threadIdx.x, lane = tid & 31, wid = tid >> 5;
  const int half = (lane >> 4) << 3, nn = lane & 15;
  const int n0 = blockIdx.x * 16;
  const int kc = blockIdx.y;
  const long kbase = (long)kc * 2048 + wid * 256;
  f32x8 acc[2] = {};
  for (int kk = 0; kk < 256; kk += 32) {
    long kg = kbase + kk;
    __builtin_prefetch(Wdt + (long)n0 * KDOWN + kg + 256, 0, 1);  // global_prefetch_b8
    bf16x16 bfr = fragA(Wdt + (long)n0 * KDOWN + kg, KDOWN, lane);
#pragma unroll
    for (int mi = 0; mi < 2; mi++) {
      bf16x16 afr = fragA(A + (long)(mi * 16) * KDOWN + kg, KDOWN, lane);
      acc[mi] = wmma_bf16(afr, bfr, acc[mi]);
    }
  }
#pragma unroll
  for (int mi = 0; mi < 2; mi++)
#pragma unroll
    for (int r = 0; r < 8; r++) red[wid][mi][half + r][nn] = acc[mi][r];
  __syncthreads();
  for (int e = tid; e < 512; e += 256) {
    int mi = e >> 8, m = (e >> 4) & 15, n = e & 15;
    float ssum = 0.f;
#pragma unroll
    for (int w = 0; w < 8; w++) ssum += red[w][mi][m][n];
    part[((long)kc * 32 + mi * 16 + m) * ODIM + n0 + n] = ssum;
  }
}

// down pass2: deterministic reduction of 64 K-chunk partials + bias.
__global__ __launch_bounds__(256) void k_down2(const float* __restrict__ part,
                                               const float* __restrict__ bias,
                                               float* __restrict__ out) {
  const int m = blockIdx.x, n = threadIdx.x;
  float s = bias[n];
  for (int kc = 0; kc < 64; kc++) s += part[((long)kc * 32 + m) * ODIM + n];
  out[(long)m * ODIM + n] = s;
}

// ---------------- host launch ----------------
extern "C" void kernel_launch(void* const* d_in, const int* in_sizes, int n_in,
                              void* d_out, int out_size, void* d_ws, size_t ws_size,
                              hipStream_t stream) {
  (void)in_sizes; (void)n_in; (void)out_size; (void)ws_size;
  const float* x      = (const float*)d_in[0];
  const int*   adj    = (const int*)  d_in[1];
  const float* upp_w  = (const float*)d_in[2];
  const float* upp_b  = (const float*)d_in[3];
  const float* wq     = (const float*)d_in[4];
  const float* bq     = (const float*)d_in[5];
  const float* wk     = (const float*)d_in[6];
  const float* bk     = (const float*)d_in[7];
  const float* wv     = (const float*)d_in[8];
  const float* bv     = (const float*)d_in[9];
  const float* fc_w   = (const float*)d_in[10];
  const float* fc_b   = (const float*)d_in[11];
  const float* ln1_g  = (const float*)d_in[12];
  const float* ln1_b  = (const float*)d_in[13];
  const float* ff1_w  = (const float*)d_in[14];
  const float* ff1_b  = (const float*)d_in[15];
  const float* ff2_w  = (const float*)d_in[16];
  const float* ff2_b  = (const float*)d_in[17];
  const float* ln2_g  = (const float*)d_in[18];
  const float* ln2_b  = (const float*)d_in[19];
  const float* down_w = (const float*)d_in[20];
  const float* down_b = (const float*)d_in[21];
  float* out = (float*)d_out;

  char* p = (char*)d_ws;
  auto take = [&](size_t bytes) -> void* {
    void* r = (void*)p; p += (bytes + 255) & ~(size_t)255; return r;
  };
  bf16*  xb     = (bf16*)take((size_t)MROWS * WDIM * 2);
  bf16*  wuppt  = (bf16*)take((size_t)WDIM * EDIM * 2);   // [E][W]
  bf16*  wqt    = (bf16*)take((size_t)EDIM * HD * 2);     // [HD][E]
  bf16*  wkt    = (bf16*)take((size_t)EDIM * HD * 2);
  bf16*  wvt    = (bf16*)take((size_t)EDIM * HD * 2);
  bf16*  wfct   = (bf16*)take((size_t)HD * EDIM * 2);     // [E][HD]
  bf16*  wff1t  = (bf16*)take((size_t)EDIM * EDIM * 2);
  bf16*  wff2t  = (bf16*)take((size_t)EDIM * EDIM * 2);
  bf16*  wdt    = (bf16*)take((size_t)KDOWN * ODIM * 2);  // [O][KDOWN]
  unsigned* adjm = (unsigned*)take((size_t)MROWS * AWORDS * 4);
  float* h0f    = (float*)take((size_t)MROWS * EDIM * 4);
  bf16*  h0b    = (bf16*) take((size_t)MROWS * EDIM * 2);
  bf16*  qb     = (bf16*) take((size_t)MROWS * HD * 2);
  bf16*  kb     = (bf16*) take((size_t)MROWS * HD * 2);
  bf16*  vb     = (bf16*) take((size_t)MROWS * HD * 2);
  bf16*  vt     = (bf16*) take((size_t)MROWS * HD * 2);   // [B][HD][N]
  bf16*  ctxb   = (bf16*) take((size_t)MROWS * HD * 2);
  float* aof    = (float*)take((size_t)MROWS * EDIM * 4);
  float* h1f    = (float*)take((size_t)MROWS * EDIM * 4);
  bf16*  h1b    = (bf16*) take((size_t)MROWS * EDIM * 2);
  bf16*  ffb    = (bf16*) take((size_t)MROWS * EDIM * 2);
  float* f2f    = (float*)take((size_t)MROWS * EDIM * 4);
  bf16*  h2b    = (bf16*) take((size_t)MROWS * EDIM * 2);
  float* dpart  = (float*)take((size_t)64 * 32 * ODIM * 4);

  // x stays row-major (A operand)
  {
    long n = (long)MROWS * WDIM;
    int blocks = (int)((n + 2047) / 2048); if (blocks > 8192) blocks = 8192;
    k_cvt<<<blocks, 256, 0, stream>>>(x, xb, n);
  }
  // pack adjacency into bit-words (32MB int32 -> 1MB bits, L2-resident)
  {
    long words = (long)MROWS * AWORDS;
    k_adjpack<<<(int)((words + 255) / 256), 256, 0, stream>>>(adj, adjm, words);
  }
  // weights converted transposed so B-fragments are contiguous global b128 loads
  dim3 tb(32, 8);
  k_cvt_t<<<dim3(WDIM / 32, EDIM / 32), tb, 0, stream>>>(upp_w,  wuppt, WDIM, EDIM);
  k_cvt_t<<<dim3(EDIM / 32, HD / 32),   tb, 0, stream>>>(wq,     wqt,   EDIM, HD);
  k_cvt_t<<<dim3(EDIM / 32, HD / 32),   tb, 0, stream>>>(wk,     wkt,   EDIM, HD);
  k_cvt_t<<<dim3(EDIM / 32, HD / 32),   tb, 0, stream>>>(wv,     wvt,   EDIM, HD);
  k_cvt_t<<<dim3(HD / 32, EDIM / 32),   tb, 0, stream>>>(fc_w,   wfct,  HD, EDIM);
  k_cvt_t<<<dim3(EDIM / 32, EDIM / 32), tb, 0, stream>>>(ff1_w,  wff1t, EDIM, EDIM);
  k_cvt_t<<<dim3(EDIM / 32, EDIM / 32), tb, 0, stream>>>(ff2_w,  wff2t, EDIM, EDIM);
  k_cvt_t<<<dim3(KDOWN / 32, ODIM / 32), tb, 0, stream>>>(down_w, wdt,  KDOWN, ODIM);

  // upp: [16384,128] x [128,256]
  k_gemm<false, true, true><<<dim3(MROWS / 128, EDIM / 128), 256, 0, stream>>>(
      xb, wuppt, upp_b, h0f, h0b, MROWS, EDIM, WDIM);
  // Q,K,V: [16384,256] x [256,512]
  k_gemm<false, false, true><<<dim3(MROWS / 128, HD / 128), 256, 0, stream>>>(
      h0b, wqt, bq, nullptr, qb, MROWS, HD, EDIM);
  k_gemm<false, false, true><<<dim3(MROWS / 128, HD / 128), 256, 0, stream>>>(
      h0b, wkt, bk, nullptr, kb, MROWS, HD, EDIM);
  k_gemm<false, false, true><<<dim3(MROWS / 128, HD / 128), 256, 0, stream>>>(
      h0b, wvt, bv, nullptr, vb, MROWS, HD, EDIM);
  // V transpose for WMMA-friendly ctx B-fragments
  k_vtrans<<<dim3(NNODE / 32, HD / 32, BB), tb, 0, stream>>>(vb, vt);
  // attention
  k_attn<<<dim3(NNODE / 128, BB * HEADS), 256, 0, stream>>>(qb, kb, vt, adjm, ctxb);
  // fc: [16384,512] x [512,256]
  k_gemm<false, true, false><<<dim3(MROWS / 128, EDIM / 128), 256, 0, stream>>>(
      ctxb, wfct, fc_b, aof, nullptr, MROWS, EDIM, HD);
  // ln1(out + residual h0)
  k_addln<<<MROWS, 256, 0, stream>>>(aof, h0f, ln1_g, ln1_b, h1f, h1b);
  // ffn
  k_gemm<true, false, true><<<dim3(MROWS / 128, EDIM / 128), 256, 0, stream>>>(
      h1b, wff1t, ff1_b, nullptr, ffb, MROWS, EDIM, EDIM);
  k_gemm<false, true, false><<<dim3(MROWS / 128, EDIM / 128), 256, 0, stream>>>(
      ffb, wff2t, ff2_b, f2f, nullptr, MROWS, EDIM, EDIM);
  k_addln<<<MROWS, 256, 0, stream>>>(f2f, h1f, ln2_g, ln2_b, nullptr, h2b);
  // down projection (K-split, deterministic two-pass)
  k_down1<<<dim3(ODIM / 16, 64), 256, 0, stream>>>(h2b, wdt, dpart);
  k_down2<<<BB, 256, 0, stream>>>(dpart, down_b, out);
}
